// MAMBA_52484500357128
// MI455X (gfx1250) — compile-verified, matrix operand order checked by reading, not run on hardware
//
#include <hip/hip_runtime.h>
#include <hip/hip_bf16.h>
#include <math.h>

// ---------------------------------------------------------------------------
// MI455X (gfx1250) implementation of the SS2D / VMamba block.
//
// Pipeline (all on `stream`, scratch carved from d_ws):
//   k0  : f32 -> bf16 converts (3 weight matrices only, 384KB total)
//   k1  : z = silu(x_in @ Win^T)                  [WMMA bf16, fp32 A inline-cvt]
//   k2  : pixel-shuffle + depthwise 3x3 + silu    [VALU, mem-bound]
//   k3  : patch-embed GEMM + bias + BN -> x_post  [WMMA bf16, gathered fp32 A]
//   k4  : 4-direction selective scan -> y_sum     [VALU scan, LDS tiles]
//   k5  : fused LayerNorm*z + out_proj GEMM       [wave LN + LDS-A WMMA]
//
// Roofline: ~3.6 GFLOP, ~70MB HBM traffic after fusion -> bandwidth bound
// (floor ~3us @ 23.3 TB/s). GEMMs use v_wmma_f32_16x16x32_bf16, K-chunks of
// 32, per-lane 16B fragment loads per the CDNA5 7.12.2 VGPR layouts.
// ---------------------------------------------------------------------------

typedef __attribute__((ext_vector_type(16))) __bf16 v16bf;
typedef __attribute__((ext_vector_type(8)))  float  v8f;

__device__ __forceinline__ float silu_f(float v)     { return v / (1.f + __expf(-v)); }
__device__ __forceinline__ float softplus_f(float v) { return v > 20.f ? v : log1pf(__expf(v)); }

// --- fragment loader: row-major bf16 matrix, K chunk k0 (mult of 32).
// Layout (ISA 7.12.2): lane = row(A)/col(B); half = lane>>4;
//   v0..3 <- K in [k0+half*8, +8)   v4..7 <- K in [k0+16+half*8, +8)
__device__ __forceinline__ v16bf load_frag_rm(const __bf16* __restrict__ M,
                                              int row, int ld, int k0, int half) {
  const __bf16* p = M + row * ld + k0 + half * 8;
  union { float4 f4; __bf16 h[8]; } u0, u1;
  u0.f4 = *reinterpret_cast<const float4*>(p);
  u1.f4 = *reinterpret_cast<const float4*>(p + 16);
  v16bf f;
#pragma unroll
  for (int j = 0; j < 8; ++j) { f[j] = u0.h[j]; f[8 + j] = u1.h[j]; }
  return f;
}

// --- same fragment, but source is fp32 row-major (convert inline)
__device__ __forceinline__ v16bf load_frag_f32rm(const float* __restrict__ M,
                                                 int row, int ld, int k0, int half) {
  const float* p = M + row * ld + k0 + half * 8;
  float4 u0 = *reinterpret_cast<const float4*>(p);
  float4 u1 = *reinterpret_cast<const float4*>(p + 4);
  float4 u2 = *reinterpret_cast<const float4*>(p + 16);
  float4 u3 = *reinterpret_cast<const float4*>(p + 20);
  v16bf f;
  f[0] = (__bf16)u0.x; f[1]  = (__bf16)u0.y; f[2]  = (__bf16)u0.z; f[3]  = (__bf16)u0.w;
  f[4] = (__bf16)u1.x; f[5]  = (__bf16)u1.y; f[6]  = (__bf16)u1.z; f[7]  = (__bf16)u1.w;
  f[8] = (__bf16)u2.x; f[9]  = (__bf16)u2.y; f[10] = (__bf16)u2.z; f[11] = (__bf16)u2.w;
  f[12] = (__bf16)u3.x; f[13] = (__bf16)u3.y; f[14] = (__bf16)u3.z; f[15] = (__bf16)u3.w;
  return f;
}

// --- k0: float -> bf16 convert (weights only)
__global__ void f2bf_kernel(const float* __restrict__ in, __bf16* __restrict__ out, int n) {
  int i = blockIdx.x * blockDim.x + threadIdx.x;
  if (i < n) out[i] = (__bf16)in[i];
}

// --- k1: z = silu(x_in(8192x256,f32) @ Win(256x256,bf16)^T)
// block = 128 thr (4 waves); wave computes one 16x16 tile; grid (512, 4).
__global__ void gemm_in_proj(const float* __restrict__ A, const __bf16* __restrict__ Bw,
                             float* __restrict__ C) {
  const int lane = threadIdx.x & 31;
  const int wave = threadIdx.x >> 5;
  const int half = lane >> 4;
  const int r    = lane & 15;
  const int m0   = blockIdx.x * 16;
  const int n0   = blockIdx.y * 64 + wave * 16;

  v8f acc = {0.f, 0.f, 0.f, 0.f, 0.f, 0.f, 0.f, 0.f};
#pragma unroll
  for (int kk = 0; kk < 256; kk += 32) {
    v16bf a = load_frag_f32rm(A, m0 + r, 256, kk, half);
    v16bf b = load_frag_rm(Bw, n0 + r, 256, kk, half);
    acc = __builtin_amdgcn_wmma_f32_16x16x32_bf16(false, a, false, b, (short)0, acc,
                                                  false, false);
  }
#pragma unroll
  for (int j = 0; j < 8; ++j) {               // D layout: VGPR j -> M = j + 8*half
    int mrow = m0 + j + 8 * half;
    int ncol = n0 + r;
    C[mrow * 256 + ncol] = silu_f(acc[j]);
  }
}

// --- k2: pixel-shuffle + depthwise 3x3 (SAME, groups=4) + bias + silu
__global__ void shuffle_dwconv(const float* __restrict__ x_in,
                               const float* __restrict__ conv_w,
                               const float* __restrict__ conv_b,
                               float* __restrict__ x_conv) {
  int idx = blockIdx.x * blockDim.x + threadIdx.x;
  if (idx >= 128 * 4 * 64 * 64) return;
  int w = idx & 63, h = (idx >> 6) & 63, c = (idx >> 12) & 3, b = idx >> 14;
  float acc = conv_b[c];
#pragma unroll
  for (int dr = -1; dr <= 1; ++dr) {
#pragma unroll
    for (int ds = -1; ds <= 1; ++ds) {
      int hh = h + dr, ww = w + ds;
      if (hh < 0 || hh > 63 || ww < 0 || ww > 63) continue;
      float v = x_in[(((b * 8 + (hh >> 3)) * 8 + (ww >> 3)) * 256)
                     + c * 64 + ((hh & 7) << 3) + (ww & 7)];
      acc += conv_w[c * 9 + (dr + 1) * 3 + (ds + 1)] * v;
    }
  }
  x_conv[idx] = silu_f(acc);
}

// --- k3: patch-embed GEMM. A[i=token, d=c*64+r*8+s] gathered from x_conv:
// each 8-wide K-run (d0 mult of 8) is 8 contiguous floats of one row.
__device__ __forceinline__ v16bf load_frag_patch(const float* __restrict__ xc,
                                                 int token, int k0, int half) {
  int b = token >> 6, l = token & 63, p = l >> 3, q = l & 7;
  v16bf f;
  int d0 = k0 + half * 8;
#pragma unroll
  for (int part = 0; part < 2; ++part) {
    int dd = d0 + part * 16;
    int c = dd >> 6, r = (dd >> 3) & 7;
    const float* src = xc + (((b * 4 + c) * 64) + p * 8 + r) * 64 + q * 8;
    float4 u0 = *reinterpret_cast<const float4*>(src);
    float4 u1 = *reinterpret_cast<const float4*>(src + 4);
    f[part * 8 + 0] = (__bf16)u0.x; f[part * 8 + 1] = (__bf16)u0.y;
    f[part * 8 + 2] = (__bf16)u0.z; f[part * 8 + 3] = (__bf16)u0.w;
    f[part * 8 + 4] = (__bf16)u1.x; f[part * 8 + 5] = (__bf16)u1.y;
    f[part * 8 + 6] = (__bf16)u1.z; f[part * 8 + 7] = (__bf16)u1.w;
  }
  return f;
}

__global__ void patch_gemm_bn(const float* __restrict__ xconv, const __bf16* __restrict__ Wp,
                              const float* __restrict__ pb, const float* __restrict__ bng,
                              const float* __restrict__ bnb, const float* __restrict__ bnm,
                              const float* __restrict__ bnv, float* __restrict__ x_post) {
  const int lane = threadIdx.x & 31;
  const int wave = threadIdx.x >> 5;
  const int half = lane >> 4;
  const int r    = lane & 15;
  const int m0   = blockIdx.x * 16;
  const int n0   = blockIdx.y * 64 + wave * 16;

  v8f acc = {0.f, 0.f, 0.f, 0.f, 0.f, 0.f, 0.f, 0.f};
#pragma unroll
  for (int kk = 0; kk < 256; kk += 32) {
    v16bf a = load_frag_patch(xconv, m0 + r, kk, half);
    v16bf b = load_frag_rm(Wp, n0 + r, 256, kk, half);
    acc = __builtin_amdgcn_wmma_f32_16x16x32_bf16(false, a, false, b, (short)0, acc,
                                                  false, false);
  }
#pragma unroll
  for (int j = 0; j < 8; ++j) {
    int token = m0 + j + 8 * half;
    int e     = n0 + r;
    int bb = token >> 6, l = token & 63;
    float v = acc[j] + pb[e];
    v = (v - bnm[e]) * (rsqrtf(bnv[e] + 1e-5f) * bng[e]) + bnb[e];
    x_post[(bb * 256 + e) * 64 + l] = v;            // (B, 256, L) layout
  }
}

// --- k4: 4-direction selective scan per (b,g). Direction maps are
// involutions => gather map == scatter map; accumulate all 4 k's in LDS.
__device__ __forceinline__ int dirmap(int k, int l) {
  switch (k) {
    case 0:  return l;
    case 1:  return ((l & 7) << 3) | (l >> 3);
    case 2:  return 63 - l;
    default: { int m = 63 - l; return ((m & 7) << 3) | (m >> 3); }
  }
}

__global__ void ssm_scan(const float* __restrict__ x_post, const float* __restrict__ x_proj_w,
                         const float* __restrict__ dt_w, const float* __restrict__ dt_b,
                         const float* __restrict__ A_logs, const float* __restrict__ Ds,
                         float* __restrict__ y_sum) {
  __shared__ float xg[64][65];     // xg[d][l]
  __shared__ float xdbl[18][64];   // rows 0..5 dts, 6..11 Bs, 12..17 Cs
  __shared__ float xpw[18 * 64];
  __shared__ float ys[64][65];     // accumulated y per (d, l')

  const int t  = threadIdx.x;                 // 0..63 == d
  const int g  = blockIdx.x & 3;
  const int b  = blockIdx.x >> 2;
  const float* xp_base = x_post + (b * 256 + g * 64) * 64;

  for (int l = 0; l < 64; ++l) ys[t][l] = 0.f;

  for (int k = 0; k < 4; ++k) {
    __syncthreads();
    const float* row = xp_base + t * 64;
    for (int l = 0; l < 64; ++l) xg[t][l] = row[dirmap(k, l)];
    const float* W = x_proj_w + (g * 4 + k) * 18 * 64;
    for (int i = t; i < 18 * 64; i += 64) xpw[i] = W[i];
    __syncthreads();

    // x_dbl[c, l=t] = sum_d xg[d, t] * xpw[c, d]
    for (int c = 0; c < 18; ++c) {
      float s = 0.f;
      const float* wc = xpw + c * 64;
      for (int d = 0; d < 64; ++d) s += xg[d][t] * wc[d];
      xdbl[c][t] = s;
    }
    __syncthreads();

    float dtw[6], a6[6], h[6];
    const float* dtwp = dt_w  + ((g * 4 + k) * 64 + t) * 6;
    const float* alp  = A_logs + ((g * 4 + k) * 64 + t) * 6;
#pragma unroll
    for (int n = 0; n < 6; ++n) { dtw[n] = dtwp[n]; a6[n] = -__expf(alp[n]); h[n] = 0.f; }
    const float dtb = dt_b[(g * 4 + k) * 64 + t];
    const float dsv = Ds  [(g * 4 + k) * 64 + t];

    for (int l = 0; l < 64; ++l) {
      float de = dtb;
#pragma unroll
      for (int rr = 0; rr < 6; ++rr) de += xdbl[rr][l] * dtw[rr];
      de = softplus_f(de);
      float xv = xg[t][l];
      float dx = de * xv;
      float yv = 0.f;
#pragma unroll
      for (int n = 0; n < 6; ++n) {
        h[n] = __expf(de * a6[n]) * h[n] + dx * xdbl[6 + n][l];
        yv  += h[n] * xdbl[12 + n][l];
      }
      yv += dsv * xv;
      ys[t][dirmap(k, l)] += yv;
    }
    __syncthreads();
  }

  float* out = y_sum + ((b * 4 + g) * 64 + t) * 64;   // (B, G, Dg, L)
  for (int l = 0; l < 64; ++l) out[l] = ys[t][l];
}

// --- k5: fused per-token LayerNorm * z  +  out_proj GEMM.
// block = 512 thr (16 waves); grid = 512 m-tiles of 16 tokens.
// Wave w: (a) LayerNorm token m0+w with wave32 shuffle reductions, write
// normalized*z row as bf16 into LDS; (b) after barrier, WMMA the 16x16
// output tile at n0 = w*16 with A-fragments read from LDS.
__global__ void ln_outproj(const float* __restrict__ y_sum, const float* __restrict__ z,
                           const float* __restrict__ ln_g, const float* __restrict__ ln_b,
                           const __bf16* __restrict__ Wout, float* __restrict__ out) {
  __shared__ __bf16 tileA[16][256];

  const int lane = threadIdx.x & 31;
  const int wave = threadIdx.x >> 5;     // 0..15
  const int m0   = blockIdx.x * 16;

  // ---- (a) LayerNorm for token m0+wave
  {
    const int token = m0 + wave;
    const int b = token >> 6, l = token & 63;
    float v[8];
#pragma unroll
    for (int j = 0; j < 8; ++j) {
      int e = j * 32 + lane;
      int g = e >> 6, d = e & 63;
      v[j] = y_sum[((b * 4 + g) * 64 + d) * 64 + l];
    }
    float s = 0.f;
#pragma unroll
    for (int j = 0; j < 8; ++j) s += v[j];
#pragma unroll
    for (int m = 16; m > 0; m >>= 1) s += __shfl_xor(s, m, 32);
    const float mu = s * (1.f / 256.f);
    float q = 0.f;
#pragma unroll
    for (int j = 0; j < 8; ++j) { float dv = v[j] - mu; q += dv * dv; }
#pragma unroll
    for (int m = 16; m > 0; m >>= 1) q += __shfl_xor(q, m, 32);
    const float rs = rsqrtf(q * (1.f / 256.f) + 1e-5f);
#pragma unroll
    for (int j = 0; j < 8; ++j) {
      int e = j * 32 + lane;
      float vn = (v[j] - mu) * rs * ln_g[e] + ln_b[e];
      vn *= z[token * 256 + e];
      tileA[wave][e] = (__bf16)vn;
    }
  }
  __syncthreads();

  // ---- (b) out[m0..+16, n0..+16] = tileA @ Wout^T
  const int half = lane >> 4;
  const int r    = lane & 15;
  const int n0   = wave * 16;
  v8f acc = {0.f, 0.f, 0.f, 0.f, 0.f, 0.f, 0.f, 0.f};
#pragma unroll
  for (int kk = 0; kk < 256; kk += 32) {
    v16bf a = load_frag_rm(&tileA[0][0], r, 256, kk, half);   // LDS ds_load_b128
    v16bf b = load_frag_rm(Wout, n0 + r, 256, kk, half);
    acc = __builtin_amdgcn_wmma_f32_16x16x32_bf16(false, a, false, b, (short)0, acc,
                                                  false, false);
  }
#pragma unroll
  for (int j = 0; j < 8; ++j) {
    int mrow = m0 + j + 8 * half;
    int ncol = n0 + r;
    out[mrow * 256 + ncol] = acc[j];
  }
}

// ---------------------------------------------------------------------------
extern "C" void kernel_launch(void* const* d_in, const int* in_sizes, int n_in,
                              void* d_out, int out_size, void* d_ws, size_t ws_size,
                              hipStream_t stream) {
  (void)in_sizes; (void)n_in; (void)out_size; (void)ws_size;
  const float* x_in      = (const float*)d_in[0];
  const float* in_proj_w = (const float*)d_in[1];
  const float* conv_w    = (const float*)d_in[2];
  const float* conv_b    = (const float*)d_in[3];
  const float* patch_w   = (const float*)d_in[4];
  const float* patch_b   = (const float*)d_in[5];
  const float* bn_gamma  = (const float*)d_in[6];
  const float* bn_beta   = (const float*)d_in[7];
  const float* bn_mean   = (const float*)d_in[8];
  const float* bn_var    = (const float*)d_in[9];
  const float* x_proj_w  = (const float*)d_in[10];
  const float* dt_w      = (const float*)d_in[11];
  const float* dt_b      = (const float*)d_in[12];
  const float* A_logs    = (const float*)d_in[13];
  const float* Ds        = (const float*)d_in[14];
  const float* ln_gamma  = (const float*)d_in[15];
  const float* ln_beta   = (const float*)d_in[16];
  const float* out_proj_w= (const float*)d_in[17];
  float* out = (float*)d_out;

  char* w = (char*)d_ws;                       // ~32.4 MB used
  __bf16* Win    = (__bf16*)(w + 0);           //    131,072
  __bf16* Wpatch = (__bf16*)(w + 131072);      //    131,072
  __bf16* Wout   = (__bf16*)(w + 262144);      //    131,072
  float*  z      = (float*) (w + 393216);      //  8,388,608
  float*  xconv  = (float*) (w + 8781824);     //  8,388,608
  float*  xpost  = (float*) (w + 17170432);    //  8,388,608
  float*  ysum   = (float*) (w + 25559040);    //  8,388,608

  // k0: weight converts (384KB total)
  f2bf_kernel<<<(65536 + 255) / 256, 256, 0, stream>>>(in_proj_w, Win, 65536);
  f2bf_kernel<<<(65536 + 255) / 256, 256, 0, stream>>>(patch_w, Wpatch, 65536);
  f2bf_kernel<<<(65536 + 255) / 256, 256, 0, stream>>>(out_proj_w, Wout, 65536);

  dim3 gemm_grid(512, 4), gemm_block(128);
  // k1: z = silu(x_in @ Win^T), fp32 A converted inline
  gemm_in_proj<<<gemm_grid, gemm_block, 0, stream>>>(x_in, Win, z);
  // k2: shuffle + depthwise conv + silu
  shuffle_dwconv<<<(2097152 + 255) / 256, 256, 0, stream>>>(x_in, conv_w, conv_b, xconv);
  // k3: patch GEMM + BN -> x_post (B,256,L)
  patch_gemm_bn<<<gemm_grid, gemm_block, 0, stream>>>(xconv, Wpatch, patch_b, bn_gamma,
                                                      bn_beta, bn_mean, bn_var, xpost);
  // k4: selective scan, 4 directions fused per (b,g)
  ssm_scan<<<512, 64, 0, stream>>>(xpost, x_proj_w, dt_w, dt_b, A_logs, Ds, ysum);
  // k5: fused LayerNorm*z + out_proj GEMM -> d_out
  ln_outproj<<<512, 512, 0, stream>>>(ysum, z, ln_gamma, ln_beta, Wout, out);
}
